// SphereTransBlock_34711925686302
// MI455X (gfx1250) — matrix operand order repaired and checked by reading, not compiled, not run on hardware
//
#include <hip/hip_runtime.h>
#include <hip/hip_bf16.h>

// ---------------------------------------------------------------------------
// SphereTransBlock for MI455X (gfx1250):
//   bf16 WMMA GEMMs (vectorized frags, transposed weights, async-LDS staging)
//   + coalesced deformable-attention gather + depthwise conv.
// ---------------------------------------------------------------------------

typedef __attribute__((ext_vector_type(16))) __bf16 v16bf;
typedef __attribute__((ext_vector_type(8)))  __bf16 v8bf;
typedef __attribute__((ext_vector_type(8)))  float  v8f;

#define BB      2
#define HH      64
#define WW      128
#define CC      256
#define HEADS   8
#define KK      9
#define HIDDEN  1024
#define HD      32
#define HWTOK   (HH * WW)          // 8192
#define NTOK    (BB * HWTOK)       // 16384

__device__ __forceinline__ float gelu_exact(float x) {
    return 0.5f * x * (1.0f + erff(x * 0.7071067811865475f));
}

__device__ __forceinline__ v16bf cat8(v8bf lo, v8bf hi) {
    return __builtin_shufflevector(lo, hi, 0, 1, 2, 3, 4, 5, 6, 7,
                                           8, 9, 10, 11, 12, 13, 14, 15);
}

// ---------------- weight conversion: fp32 [K,N] -> bf16 transposed [N,K] --
__global__ void cvt_transpose_bf16(const float* __restrict__ s, __bf16* __restrict__ d,
                                   int K, int N) {
    int i = blockIdx.x * blockDim.x + threadIdx.x;
    if (i < K * N) {
        int k = i / N, n = i - k * N;
        d[(size_t)n * K + k] = (__bf16)s[i];
    }
}

// w_attn [256,72] -> bf16 transposed padded [80,256]; bias [72] -> fp32 [80]
__global__ void pad_attn_wT(const float* __restrict__ w, const float* __restrict__ b,
                            __bf16* __restrict__ wd, float* __restrict__ bd) {
    int i = blockIdx.x * blockDim.x + threadIdx.x;
    if (i < 80 * 256) {
        int n = i >> 8, k = i & 255;
        wd[i] = (n < 72) ? (__bf16)w[k * 72 + n] : (__bf16)0.0f;
    }
    if (i < 80) bd[i] = (i < 72) ? b[i] : 0.0f;
}

// ---------------- LayerNorm (one wave32 per token, C=256) ----------------
__global__ __launch_bounds__(256)
void layernorm_bf16(const float* __restrict__ xin, const float* __restrict__ g,
                    const float* __restrict__ bta, __bf16* __restrict__ out, int ntok) {
    int gtid  = blockIdx.x * blockDim.x + threadIdx.x;
    int token = gtid >> 5;
    int lane  = gtid & 31;
    if (token >= ntok) return;
    const float* row = xin + (size_t)token * CC;
    float v[8];
    float s = 0.0f;
#pragma unroll
    for (int i = 0; i < 8; ++i) { v[i] = row[lane + i * 32]; s += v[i]; }
#pragma unroll
    for (int m = 16; m >= 1; m >>= 1) s += __shfl_xor(s, m, 32);
    float mean = s * (1.0f / 256.0f);
    float var = 0.0f;
#pragma unroll
    for (int i = 0; i < 8; ++i) { float d = v[i] - mean; var += d * d; }
#pragma unroll
    for (int m = 16; m >= 1; m >>= 1) var += __shfl_xor(var, m, 32);
    var *= (1.0f / 256.0f);
    float rinv = rsqrtf(var + 1e-5f);
#pragma unroll
    for (int i = 0; i < 8; ++i) {
        int c = lane + i * 32;
        out[(size_t)token * CC + c] = (__bf16)((v[i] - mean) * rinv * g[c] + bta[c]);
    }
}

// ---------------- WMMA GEMM, async-LDS B staging (N % 64 == 0) -----------
// C[M,N] = act(A[M,K] @ B[K,N] + bias) + resid ; Bt is B transposed [N,K] bf16.
// Block = 8 waves = 128x64 tile. Each wave: 16 rows x 4 accumulator tiles.
// B panel (32K x 64N) staged in LDS per k-step via global_load_async_to_lds.
__global__ __launch_bounds__(256)
void gemm_bf16_wmma_lds(const __bf16* __restrict__ A, int lda,
                        const __bf16* __restrict__ Bt,
                        const float* __restrict__ bias,
                        const float* __restrict__ resid,
                        float* __restrict__ outF, __bf16* __restrict__ outB,
                        int M, int N, int Kdim, int actGelu) {
    __shared__ __align__(16) __bf16 sB[64 * 32];   // [col][K] : 4 KB
    int wave = threadIdx.x >> 5;
    int lane = threadIdx.x & 31;
    int nGroups = N >> 6;
    int bm = blockIdx.x / nGroups;
    int bn = blockIdx.x - bm * nGroups;
    int g  = lane >> 4;
    int mr = lane & 15;

    const __bf16* Arow = A + (size_t)(bm * 128 + wave * 16 + mr) * lda;

    // async-copy role: thread t copies 8 K-elements (16 B) of one column
    int ccol  = threadIdx.x >> 2;      // 0..63
    int cpart = threadIdx.x & 3;       // 0..3
    const __bf16* Bsrc = Bt + (size_t)(bn * 64 + ccol) * Kdim + cpart * 8;
    unsigned ldsDst = (unsigned)(size_t)(&sB[ccol * 32 + cpart * 8]);

    v8f acc[4] = {};
    for (int k0 = 0; k0 < Kdim; k0 += 32) {
        __syncthreads();               // previous panel fully consumed
        {
            unsigned long long ga = (unsigned long long)(Bsrc + k0);
            asm volatile("global_load_async_to_lds_b128 %0, %1, off"
                         :: "v"(ldsDst), "v"(ga) : "memory");
        }
        // A fragment from global (two 16-B vector loads, k = 8g..8g+7 then 16+8g..16+8g+7)
        v8bf alo = *(const v8bf*)(Arow + k0 + g * 8);
        v8bf ahi = *(const v8bf*)(Arow + k0 + 16 + g * 8);
        v16bf a = cat8(alo, ahi);
        __builtin_prefetch(Arow + k0 + 32, 0, 0);   // global_prefetch_b8
        asm volatile("s_wait_asynccnt 0" ::: "memory");
        __syncthreads();               // panel visible to all waves
#pragma unroll
        for (int j = 0; j < 4; ++j) {
            const __bf16* bp = &sB[(j * 16 + mr) * 32 + g * 8];
            v16bf b = cat8(*(const v8bf*)bp, *(const v8bf*)(bp + 16));
            acc[j] = __builtin_amdgcn_wmma_f32_16x16x32_bf16(
                false, a, false, b, (short)0, acc[j], false, false);
        }
    }
#pragma unroll
    for (int j = 0; j < 4; ++j) {
        int col = bn * 64 + j * 16 + mr;
#pragma unroll
        for (int r = 0; r < 8; ++r) {
            int row = bm * 128 + wave * 16 + r + 8 * g;
            float t = acc[j][r];
            if (bias) t += bias[col];
            if (actGelu) t = gelu_exact(t);
            size_t idx = (size_t)row * N + col;
            if (resid) t += resid[idx];
            if (outF) outF[idx] = t;
            if (outB) outB[idx] = (__bf16)t;
        }
    }
}

// ---------------- WMMA GEMM, direct loads (odd N: 144 -> TN=3, 80 -> TN=5)
template <int TN>
__global__ __launch_bounds__(256)
void gemm_bf16_wmma_direct(const __bf16* __restrict__ A, int lda,
                           const __bf16* __restrict__ Bt,
                           const float* __restrict__ bias,
                           float* __restrict__ outF,
                           int M, int N, int Kdim) {
    int wave = threadIdx.x >> 5;
    int lane = threadIdx.x & 31;
    int ntiles = N >> 4;
    int groups = ntiles / TN;
    int total  = (M >> 4) * groups;
    int tid = blockIdx.x * 8 + wave;
    if (tid >= total) return;          // wave-uniform
    int tm = tid / groups;
    int tg = tid - tm * groups;
    int g  = lane >> 4;
    int mr = lane & 15;

    const __bf16* Arow = A + (size_t)(tm * 16 + mr) * lda;
    v8f acc[TN] = {};
    for (int k0 = 0; k0 < Kdim; k0 += 32) {
        v16bf a = cat8(*(const v8bf*)(Arow + k0 + g * 8),
                       *(const v8bf*)(Arow + k0 + 16 + g * 8));
#pragma unroll
        for (int j = 0; j < TN; ++j) {
            const __bf16* Brow = Bt + (size_t)((tg * TN + j) * 16 + mr) * Kdim + k0 + g * 8;
            v16bf b = cat8(*(const v8bf*)Brow, *(const v8bf*)(Brow + 16));
            acc[j] = __builtin_amdgcn_wmma_f32_16x16x32_bf16(
                false, a, false, b, (short)0, acc[j], false, false);
        }
    }
#pragma unroll
    for (int j = 0; j < TN; ++j) {
        int col = (tg * TN + j) * 16 + mr;
#pragma unroll
        for (int r = 0; r < 8; ++r) {
            int row = tm * 16 + r + 8 * g;
            float t = acc[j][r] + (bias ? bias[col] : 0.0f);
            outF[(size_t)row * N + col] = t;
        }
    }
}

// ---------------- Deformable attention: softmax + bilinear + aggregate ---
__global__ __launch_bounds__(256)
void deform_attn(const float* __restrict__ vbuf,     // [NTOK, 256]
                 const float* __restrict__ offbuf,   // [NTOK, 144]
                 const float* __restrict__ attnbuf,  // [NTOK, 80] (72 valid)
                 const float* __restrict__ refp,     // [H,W,K,2]
                 __bf16* __restrict__ aggbf) {       // [NTOK, 256]
    int token = blockIdx.x;
    int head  = threadIdx.x >> 5;
    int d     = threadIdx.x & 31;
    int b  = token >> 13;
    int hw = token & (HWTOK - 1);

    const float* al = attnbuf + (size_t)token * 80 + head * KK;
    float lg[KK];
    float mx = -1e30f;
#pragma unroll
    for (int k = 0; k < KK; ++k) { lg[k] = al[k]; mx = fmaxf(mx, lg[k]); }
    float sum = 0.0f;
#pragma unroll
    for (int k = 0; k < KK; ++k) { lg[k] = __expf(lg[k] - mx); sum += lg[k]; }
    float rs = 1.0f / sum;

    const float* offp = offbuf + (size_t)token * 144 + head * (KK * 2);
    const float* rp   = refp + (size_t)hw * (KK * 2);
    size_t vbase = ((size_t)b * HWTOK) * CC + head * HD + d;

    float accv = 0.0f;
#pragma unroll
    for (int k = 0; k < KK; ++k) {
        float px = rp[k * 2 + 0] + offp[k * 2 + 0];
        float py = rp[k * 2 + 1] + offp[k * 2 + 1];
        float cx = (px + 1.0f) * 0.5f * (float)(WW - 1);
        float cy = (py + 1.0f) * 0.5f * (float)(HH - 1);
        float fx = floorf(cx), fy = floorf(cy);
        float wx = cx - fx,   wy = cy - fy;
        int x0 = min(max((int)fx, 0), WW - 1);
        int x1 = min(max((int)fx + 1, 0), WW - 1);
        int y0 = min(max((int)fy, 0), HH - 1);
        int y1 = min(max((int)fy + 1, 0), HH - 1);
        float s00 = vbuf[vbase + (size_t)(y0 * WW + x0) * CC];
        float s01 = vbuf[vbase + (size_t)(y0 * WW + x1) * CC];
        float s10 = vbuf[vbase + (size_t)(y1 * WW + x0) * CC];
        float s11 = vbuf[vbase + (size_t)(y1 * WW + x1) * CC];
        float sv = s00 * (1.f - wy) * (1.f - wx) + s01 * (1.f - wy) * wx
                 + s10 * wy * (1.f - wx)         + s11 * wy * wx;
        accv += sv * lg[k] * rs;
    }
    aggbf[(size_t)token * CC + head * HD + d] = (__bf16)accv;
}

// ---------------- Depthwise 3x3 (circular W, zero H) + GELU --------------
__global__ __launch_bounds__(256)
void dwconv_gelu(const __bf16* __restrict__ h, const float* __restrict__ wdw,
                 const float* __restrict__ bdw, __bf16* __restrict__ out) {
    size_t idx = (size_t)blockIdx.x * 256 + threadIdx.x;  // B*HW*HIDDEN
    int c = (int)(idx & (HIDDEN - 1));
    size_t t = idx >> 10;
    int b  = (int)(t >> 13);
    int hw = (int)(t & (HWTOK - 1));
    int y = hw >> 7, x = hw & (WW - 1);
    float acc = 0.0f;
#pragma unroll
    for (int dy = 0; dy < 3; ++dy) {
        int yy = y + dy - 1;
        if (yy < 0 || yy >= HH) continue;     // zero pad along height
#pragma unroll
        for (int dx = 0; dx < 3; ++dx) {
            int xx = (x + dx - 1 + WW) & (WW - 1);   // circular along width
            float hv = (float)h[(((size_t)b * HWTOK + yy * WW + xx) << 10) + c];
            acc += hv * wdw[c * 9 + dy * 3 + dx];
        }
    }
    acc = gelu_exact(acc + bdw[c]);
    out[idx] = (__bf16)acc;
}

// ---------------------------------------------------------------------------
extern "C" void kernel_launch(void* const* d_in, const int* in_sizes, int n_in,
                              void* d_out, int out_size, void* d_ws, size_t ws_size,
                              hipStream_t stream) {
    const float* x      = (const float*)d_in[0];
    const float* refp   = (const float*)d_in[1];
    const float* ln1_g  = (const float*)d_in[2];
    const float* ln1_b  = (const float*)d_in[3];
    const float* w_v    = (const float*)d_in[4];
    const float* b_v    = (const float*)d_in[5];
    const float* w_off  = (const float*)d_in[6];
    const float* b_off  = (const float*)d_in[7];
    const float* w_attn = (const float*)d_in[8];
    const float* b_attn = (const float*)d_in[9];
    const float* w_out  = (const float*)d_in[10];
    const float* b_out  = (const float*)d_in[11];
    const float* ln2_g  = (const float*)d_in[12];
    const float* ln2_b  = (const float*)d_in[13];
    const float* w1     = (const float*)d_in[14];
    const float* b1     = (const float*)d_in[15];
    const float* w_dw   = (const float*)d_in[16];
    const float* b_dw   = (const float*)d_in[17];
    const float* w2     = (const float*)d_in[18];
    const float* b2     = (const float*)d_in[19];
    float* out = (float*)d_out;
    char* ws = (char*)d_ws;

    // --- workspace layout (bytes); all weight buffers hold B^T [N,K] bf16 ---
    __bf16* wvT   = (__bf16*)(ws + 0);        // [256,256]  131072
    __bf16* woffT = (__bf16*)(ws + 131072);   // [144,256]  73728
    __bf16* watT  = (__bf16*)(ws + 204800);   // [80,256]   40960
    float*  batP  = (float*) (ws + 245760);   // [80]       512
    __bf16* woutT = (__bf16*)(ws + 246272);   // [256,256]  131072
    __bf16* w1T   = (__bf16*)(ws + 377344);   // [1024,256] 524288
    __bf16* w2T   = (__bf16*)(ws + 901632);   // [256,1024] 524288
    const size_t S = 2ull * 1024 * 1024;
    __bf16* xn_bf  = (__bf16*)(ws + S);                  //  8 MB
    float*  vbuf   = (float*) (ws + S + 8388608);        // 16 MB
    float*  offbuf = (float*) (ws + S + 25165824);       // 9.4 MB
    float*  atbuf  = (float*) (ws + S + 34603008);       // 5.2 MB
    __bf16* agg_bf = (__bf16*)(ws + S + 39845888);       //  8 MB
    __bf16* y_bf   = (__bf16*)(ws + S);                  // reuse xn
    __bf16* h_bf   = (__bf16*)(ws + S + 8388608);        // reuse (32 MB)
    __bf16* h2_bf  = (__bf16*)(ws + S + 48234496);       // 32 MB

    // --- weight conversions (transpose to [N,K]) ---
    cvt_transpose_bf16<<<(65536 + 255) / 256, 256, 0, stream>>>(w_v, wvT, 256, 256);
    cvt_transpose_bf16<<<(36864 + 255) / 256, 256, 0, stream>>>(w_off, woffT, 256, 144);
    cvt_transpose_bf16<<<(65536 + 255) / 256, 256, 0, stream>>>(w_out, woutT, 256, 256);
    cvt_transpose_bf16<<<(262144 + 255) / 256, 256, 0, stream>>>(w1, w1T, 256, 1024);
    cvt_transpose_bf16<<<(262144 + 255) / 256, 256, 0, stream>>>(w2, w2T, 1024, 256);
    pad_attn_wT<<<(20480 + 255) / 256, 256, 0, stream>>>(w_attn, b_attn, watT, batP);

    // --- LN1 -> bf16 ---
    layernorm_bf16<<<(NTOK * 32) / 256, 256, 0, stream>>>(x, ln1_g, ln1_b, xn_bf, NTOK);

    // --- projections ---
    // v = xn @ w_v + b_v : [16384 x 256], K=256  (128x64 blocks)
    gemm_bf16_wmma_lds<<<(NTOK / 128) * (CC / 64), 256, 0, stream>>>(
        xn_bf, CC, wvT, b_v, nullptr, vbuf, nullptr, NTOK, CC, CC, 0);
    // off = xn @ w_off + b_off : [16384 x 144]
    gemm_bf16_wmma_direct<3><<<((NTOK / 16) * 3 + 7) / 8, 256, 0, stream>>>(
        xn_bf, CC, woffT, b_off, offbuf, NTOK, 144, CC);
    // attn logits : [16384 x 80]
    gemm_bf16_wmma_direct<5><<<((NTOK / 16) * 1 + 7) / 8, 256, 0, stream>>>(
        xn_bf, CC, watT, batP, atbuf, NTOK, 80, CC);

    // --- softmax + bilinear sampling + aggregate ---
    deform_attn<<<NTOK, 256, 0, stream>>>(vbuf, offbuf, atbuf, refp, agg_bf);

    // --- x1 = x + agg @ w_out + b_out -> d_out ---
    gemm_bf16_wmma_lds<<<(NTOK / 128) * (CC / 64), 256, 0, stream>>>(
        agg_bf, CC, woutT, b_out, x, out, nullptr, NTOK, CC, CC, 0);

    // --- LeFF branch ---
    layernorm_bf16<<<(NTOK * 32) / 256, 256, 0, stream>>>(out, ln2_g, ln2_b, y_bf, NTOK);
    // h = gelu(y @ w1 + b1) -> bf16 : [16384 x 1024], K=256
    gemm_bf16_wmma_lds<<<(NTOK / 128) * (HIDDEN / 64), 256, 0, stream>>>(
        y_bf, CC, w1T, b1, nullptr, nullptr, h_bf, NTOK, HIDDEN, CC, 1);
    // depthwise conv + gelu -> bf16
    dwconv_gelu<<<(NTOK * HIDDEN) / 256, 256, 0, stream>>>(h_bf, w_dw, b_dw, h2_bf);
    // out = x1 + h2 @ w2 + b2 : [16384 x 256], K=1024
    gemm_bf16_wmma_lds<<<(NTOK / 128) * (CC / 64), 256, 0, stream>>>(
        h2_bf, HIDDEN, w2T, b2, out, out, nullptr, NTOK, CC, HIDDEN, 0);
}
